// MultiHeadAttention_56513179681139
// MI455X (gfx1250) — matrix operand order repaired and checked by reading, not compiled, not run on hardware
//
#include <hip/hip_runtime.h>
#include <hip/hip_bf16.h>
#include <stdint.h>

// Problem constants (Transformer-XL rel-attention reference)
#define B_   2
#define S_   1024
#define M_   1024
#define KV_  2048      // M + S
#define H_   16
#define DK_  64
#define D_   1024

typedef __attribute__((ext_vector_type(16))) __bf16 v16bf;
typedef __attribute__((ext_vector_type(8)))  __bf16 v8bf;
typedef __attribute__((ext_vector_type(8)))  float  v8f;

// ---------- small helpers ----------
static __device__ __forceinline__ unsigned short f2bf(float f) {
    unsigned u = __builtin_bit_cast(unsigned, f);
    u += 0x7fffu + ((u >> 16) & 1u);            // round-to-nearest-even
    return (unsigned short)(u >> 16);
}
static __device__ __forceinline__ float bf2f(unsigned short h) {
    unsigned u = ((unsigned)h) << 16;
    return __builtin_bit_cast(float, u);
}
static __device__ __forceinline__ v8bf ld8(const unsigned short* p) {
    return *reinterpret_cast<const v8bf*>(p);   // 16B -> global_load_b128 / ds_load_b128
}
static __device__ __forceinline__ v16bf mk16(v8bf lo, v8bf hi) {
    v16bf r;
#pragma unroll
    for (int i = 0; i < 8; ++i) { r[i] = lo[i]; r[i + 8] = hi[i]; }
    return r;
}
static __device__ __forceinline__ v8f wmma_bf16(v16bf a, v16bf b, v8f c) {
    return __builtin_amdgcn_wmma_f32_16x16x32_bf16(false, a, false, b, (short)0, c, false, false);
}
static __device__ __forceinline__ void storeC(unsigned short* p, float v) { *p = f2bf(v); }
static __device__ __forceinline__ void storeC(float* p, float v)          { *p = v; }

// ---------- CDNA5 async LDS staging (ISA §15.18.3 opcode 98, tracked by ASYNCcnt) ----------
// One instruction moves 16B per lane from global memory directly into LDS,
// bypassing VGPRs. LDS byte address comes from the VDST VGPR (low 32 bits of
// the generic shared-memory pointer, per the ISA aperture truncation rule).
static __device__ __forceinline__ void async_copy16(unsigned lds_addr, const void* gaddr) {
    asm volatile("global_load_async_to_lds_b128 %0, %1, off"
                 :: "v"(lds_addr), "v"(gaddr)
                 : "memory");
}
static __device__ __forceinline__ void wait_async_le(int n) {
    if (n == 0) asm volatile("s_wait_asynccnt 0x0" ::: "memory");
    else        asm volatile("s_wait_asynccnt 0x1" ::: "memory");
}

// ---------- elementwise kernels ----------
__global__ void k_cvt(const float* __restrict__ s, unsigned short* __restrict__ d, int n) {
    int i = blockIdx.x * blockDim.x + threadIdx.x;
    if (i < n) d[i] = f2bf(s[i]);
}
// kv = concat(memory, x) along sequence, cast to bf16.  n = B*KV*D
__global__ void k_build_kv(const float* __restrict__ mem, const float* __restrict__ x,
                           unsigned short* __restrict__ kv) {
    int i = blockIdx.x * blockDim.x + threadIdx.x;
    int b = i >> 21;                    // KV*D = 2^21
    int rem = i & ((1 << 21) - 1);
    int j = rem >> 10;                  // D = 2^10
    int c = rem & 1023;
    float v = (j < M_) ? mem[((size_t)b * M_ + j) * D_ + c]
                       : x  [((size_t)b * S_ + (j - M_)) * D_ + c];
    kv[i] = f2bf(v);
}
// q_ac = qw + ru[head,col], q_bd = qw + rv[head,col]  (col index c already == h*64+d)
__global__ void k_addbias(const unsigned short* __restrict__ qw,
                          const float* __restrict__ ru, const float* __restrict__ rv,
                          unsigned short* __restrict__ qac, unsigned short* __restrict__ qbd, int n) {
    int i = blockIdx.x * blockDim.x + threadIdx.x;
    if (i < n) {
        int c = i & 1023;
        float f = bf2f(qw[i]);
        qac[i] = f2bf(f + ru[c]);
        qbd[i] = f2bf(f + rv[c]);
    }
}
// vt[b, h, d, j] = vw[b, j, h*64+d]  -> key-contiguous rows for B-fragments of P@V
__global__ void k_transpose_v(const unsigned short* __restrict__ vw, unsigned short* __restrict__ vt) {
    int i = blockIdx.x * blockDim.x + threadIdx.x;   // n = B*KV*D
    int b = i >> 21;
    int rem = i & ((1 << 21) - 1);
    int j = rem >> 10;
    int c = rem & 1023;                              // c = h*64 + d
    vt[((size_t)(b * D_ + c)) * KV_ + j] = vw[i];
}

// ---------- bf16 GEMM:  C[M,N] = A[M,K] @ W[N,K]^T  (both operands K-contiguous) ----------
// block: 256 thr = 8 waves; block tile 128(M) x 64(N); wave tile 16 x 64.
// The 64x32 W-tile is staged into LDS ONCE per block with async-to-LDS DMA
// (double-buffered, overlapping the WMMAs), instead of each of the 8 waves
// re-fetching it from global.
template <typename OT>
__global__ __launch_bounds__(256) void gemm_bf16nt(const unsigned short* __restrict__ A,
                                                   const unsigned short* __restrict__ W,
                                                   OT* __restrict__ C,
                                                   int Mrows, int N, int K) {
    __shared__ __align__(16) unsigned short smemW[2][64 * 32];   // 2 x 4KB

    const int tid   = threadIdx.x;
    const int wid   = tid >> 5;
    const int lane  = tid & 31;
    const int lrow  = lane & 15;
    const int lhalf = lane >> 4;
    const int row0  = blockIdx.x * 128 + wid * 16;
    const int col0  = blockIdx.y * 64;
    const int kb    = lhalf ? 8 : 0;     // unused: A and B fragments both take K half kh
    const int kh    = lhalf ? 16 : 0;    // B-fragment K half

    // staging map: thread -> (row = tid/4, 16B chunk = tid%4); 256 thr cover 64x32 bf16
    const int srow   = tid >> 2;
    const int schunk = tid & 3;
    const unsigned short* Wsrc = W + (size_t)(col0 + srow) * K + schunk * 8;
    const unsigned lds0   = (unsigned)(size_t)(&smemW[0][0]);
    const unsigned lds1   = (unsigned)(size_t)(&smemW[1][0]);
    const unsigned ldsoff = (unsigned)(srow * 64 + schunk * 16);   // bytes

    async_copy16(lds0 + ldsoff, Wsrc);            // prefetch k = 0 tile

    v8f acc[4] = {};
    const unsigned short* Arow = A + (size_t)(row0 + lrow) * K;

    for (int k = 0; k < K; k += 32) {
        const int cur = (k >> 5) & 1;
        if (k + 32 < K) {
            async_copy16((cur ? lds0 : lds1) + ldsoff, Wsrc + k + 32);  // next tile in flight
            wait_async_le(1);                     // own part of current tile landed
        } else {
            wait_async_le(0);
        }
        __syncthreads();                          // whole current tile visible block-wide

        v16bf af = mk16(ld8(Arow + k + kh), ld8(Arow + k + kh + 8));
        __builtin_prefetch(Arow + k + 64, 0, 1);
#pragma unroll
        for (int nt = 0; nt < 4; ++nt) {
            const unsigned short* Wp = &smemW[cur][(nt * 16 + lrow) * 32 + kh];
            v16bf bfr = mk16(ld8(Wp), ld8(Wp + 8));
            acc[nt] = wmma_bf16(af, bfr, acc[nt]);
        }
        __syncthreads();                          // all waves done before tile is overwritten
    }
#pragma unroll
    for (int nt = 0; nt < 4; ++nt)
#pragma unroll
        for (int v = 0; v < 8; ++v) {
            int r = row0 + v + 8 * lhalf;          // C/D layout: row = vgpr + 8*(lane>=16)
            int c = col0 + nt * 16 + lrow;
            storeC(&C[(size_t)r * N + c], acc[nt][v]);
        }
}

// ---------- fused rel-pos flash attention ----------
// grid: B*H*(S/128) blocks, 256 thr (8 waves); each wave owns 16 query rows.
__global__ __launch_bounds__(256) void attn_relpos(const unsigned short* __restrict__ qac,
                                                   const unsigned short* __restrict__ qbd,
                                                   const unsigned short* __restrict__ kw,
                                                   const unsigned short* __restrict__ rw,
                                                   const unsigned short* __restrict__ vt,
                                                   const int* __restrict__ mask,
                                                   unsigned short* __restrict__ outp) {
    __shared__ __align__(16) char smem[8 * 3072];   // per wave: 2KB BD band (f32) + 1KB P (bf16)
    const int wid   = threadIdx.x >> 5;
    const int lane  = threadIdx.x & 31;
    const int lrow  = lane & 15;
    const int lhalf = lane >> 4;
    float*          lds_bd = (float*)(smem + wid * 3072);
    unsigned short* lds_p  = (unsigned short*)(smem + wid * 3072 + 2048);

    const int bid = blockIdx.x;
    const int b   = bid / (H_ * (S_ / 128));
    const int rem = bid % (H_ * (S_ / 128));
    const int h   = rem / (S_ / 128);
    const int i0  = (rem % (S_ / 128)) * 128 + wid * 16;

    const int kb = lhalf ? 8 : 0;
    const int kh = lhalf ? 16 : 0;

    // A-fragments of (q+ru) and (q+rv): 16 rows x 64 d -> two K=32 fragments each
    size_t qoff = ((size_t)(b * S_ + i0 + lrow)) * D_ + h * DK_;
    v16bf qa0 = mk16(ld8(qac + qoff + kh),      ld8(qac + qoff + kh + 8));
    v16bf qa1 = mk16(ld8(qac + qoff + 32 + kh), ld8(qac + qoff + 32 + kh + 8));
    v16bf qb0 = mk16(ld8(qbd + qoff + kh),      ld8(qbd + qoff + kh + 8));
    v16bf qb1 = mk16(ld8(qbd + qoff + 32 + kh), ld8(qbd + qoff + 32 + kh + 8));

    float mrow[8], lsum[8];
#pragma unroll
    for (int v = 0; v < 8; ++v) { mrow[v] = -1e30f; lsum[v] = 0.f; }
    v8f acc[4] = {};

    const unsigned short* kwB = kw + (size_t)b * KV_ * D_ + h * DK_;
    const unsigned short* vtB = vt + (size_t)((b * H_ + h) * DK_) * KV_;

#pragma unroll 1
    for (int j0 = 0; j0 < KV_; j0 += 32) {
        float s[2][8];
#pragma unroll
        for (int t = 0; t < 2; ++t) {
            const int jt = j0 + 16 * t;
            // ---- AC tile: (q+ru) @ K^T over d_k=64 (2 WMMAs) ----
            const unsigned short* kp = kwB + (size_t)(jt + lrow) * D_ + kh;
            v8f c8 = {};
            c8 = wmma_bf16(qa0, mk16(ld8(kp), ld8(kp + 8)), c8);
            c8 = wmma_bf16(qa1, mk16(ld8(kp + 32), ld8(kp + 40)), c8);

            // ---- BD raw band [bbase, bbase+32): shifted BD[i][j] = raw[i][j + S-1-i] ----
            const int bbase = jt + (S_ - 16) - i0;   // >= 0 always
            v8f r0 = {}, r1 = {};
            {
                int rr = bbase + lrow; rr = rr < (KV_ - 1) ? rr : (KV_ - 1);
                const unsigned short* rp = rw + (size_t)rr * D_ + h * DK_ + kh;
                r0 = wmma_bf16(qb0, mk16(ld8(rp), ld8(rp + 8)), r0);
                r0 = wmma_bf16(qb1, mk16(ld8(rp + 32), ld8(rp + 40)), r0);
            }
            {
                int rr = bbase + 16 + lrow; rr = rr < (KV_ - 1) ? rr : (KV_ - 1);
                const unsigned short* rp = rw + (size_t)rr * D_ + h * DK_ + kh;
                r1 = wmma_bf16(qb0, mk16(ld8(rp), ld8(rp + 8)), r1);
                r1 = wmma_bf16(qb1, mk16(ld8(rp + 32), ld8(rp + 40)), r1);
            }
            // stage the 16x32 raw band in wave-private LDS, then diagonal gather
#pragma unroll
            for (int v = 0; v < 8; ++v) {
                int ii = v + 8 * lhalf;
                lds_bd[ii * 32 + lrow]      = r0[v];
                lds_bd[ii * 32 + 16 + lrow] = r1[v];
            }
#pragma unroll
            for (int v = 0; v < 8; ++v) {
                int ii = v + 8 * lhalf;
                float bd = lds_bd[ii * 32 + (lrow + 15 - ii)];   // idx = jl + 15 - ii in [0,30]
                float sc = c8[v] + bd;
                int gi = i0 + ii, gj = jt + lrow;
                sc = mask[(size_t)gi * KV_ + gj] ? sc : -1e9f;
                s[t][v] = sc * 0.125f;                           // 1/sqrt(d_k)
            }
        }

        // ---- online softmax over this 32-key slab ----
#pragma unroll
        for (int v = 0; v < 8; ++v) {
            float mx = fmaxf(s[0][v], s[1][v]);
#pragma unroll
            for (int off = 1; off < 16; off <<= 1) mx = fmaxf(mx, __shfl_xor(mx, off, 32));
            float mn = fmaxf(mrow[v], mx);
            float sc = __expf(mrow[v] - mn);
            float p0 = __expf(s[0][v] - mn);
            float p1 = __expf(s[1][v] - mn);
            float rs = p0 + p1;
#pragma unroll
            for (int off = 1; off < 16; off <<= 1) rs += __shfl_xor(rs, off, 32);
            lsum[v] = lsum[v] * sc + rs;
            mrow[v] = mn;
#pragma unroll
            for (int nt = 0; nt < 4; ++nt) acc[nt][v] *= sc;
            int ii = v + 8 * lhalf;
            lds_p[ii * 32 + lrow]      = f2bf(p0);
            lds_p[ii * 32 + 16 + lrow] = f2bf(p1);
        }

        // ---- P @ V: reload P as an A-fragment (C-layout -> A-layout via LDS) ----
        const unsigned short* pp = lds_p + lrow * 32 + kh;
        v16bf pf = mk16(ld8(pp), ld8(pp + 8));
#pragma unroll
        for (int nt = 0; nt < 4; ++nt) {
            const unsigned short* vp = vtB + (size_t)(nt * 16 + lrow) * KV_ + j0 + kh;
            acc[nt] = wmma_bf16(pf, mk16(ld8(vp), ld8(vp + 8)), acc[nt]);
        }
    }

    // epilogue: normalize and store bf16 attention output [B,S,H*DK]
#pragma unroll
    for (int nt = 0; nt < 4; ++nt)
#pragma unroll
        for (int v = 0; v < 8; ++v) {
            int ii = v + 8 * lhalf;
            float o = acc[nt][v] / lsum[v];
            outp[((size_t)(b * S_ + i0 + ii)) * D_ + h * DK_ + nt * 16 + lrow] = f2bf(o);
        }
}

// ---------- host-side launcher ----------
extern "C" void kernel_launch(void* const* d_in, const int* in_sizes, int n_in,
                              void* d_out, int out_size, void* d_ws, size_t ws_size,
                              hipStream_t stream) {
    const float* x   = (const float*)d_in[0];
    const float* r   = (const float*)d_in[1];
    const float* ru  = (const float*)d_in[2];
    const float* rv  = (const float*)d_in[3];
    const int*   msk = (const int*)  d_in[4];
    const float* mem = (const float*)d_in[5];
    const float* Wq  = (const float*)d_in[6];
    const float* Wk  = (const float*)d_in[7];
    const float* Wv  = (const float*)d_in[8];
    const float* Wr  = (const float*)d_in[9];
    const float* Wo  = (const float*)d_in[10];
    float* out = (float*)d_out;

    char* ws = (char*)d_ws;
    size_t off = 0;
    auto take = [&](size_t elems) -> unsigned short* {
        unsigned short* p = (unsigned short*)(ws + off);
        off += (elems * 2 + 255) & ~(size_t)255;
        return p;
    };
    const size_t nX  = (size_t)B_ * S_ * D_;    // 2M
    const size_t nKV = (size_t)B_ * KV_ * D_;   // 4M
    const size_t nR  = (size_t)KV_ * D_;        // 2M
    const size_t nW  = (size_t)D_ * D_;         // 1M

    unsigned short* xb  = take(nX);
    unsigned short* kvb = take(nKV);
    unsigned short* rb  = take(nR);
    unsigned short* Wqb = take(nW);
    unsigned short* Wkb = take(nW);
    unsigned short* Wvb = take(nW);
    unsigned short* Wrb = take(nW);
    unsigned short* Wob = take(nW);
    unsigned short* qwb = take(nX);
    unsigned short* qac = take(nX);
    unsigned short* qbd = take(nX);
    unsigned short* kwb = take(nKV);
    unsigned short* vwb = take(nKV);
    unsigned short* vtb = take(nKV);
    unsigned short* rwb = take(nR);
    unsigned short* atb = take(nX);

    const int T = 256;
    k_cvt<<<(unsigned)((nX + T - 1) / T), T, 0, stream>>>(x, xb, (int)nX);
    k_cvt<<<(unsigned)((nR + T - 1) / T), T, 0, stream>>>(r, rb, (int)nR);
    k_cvt<<<(unsigned)((nW + T - 1) / T), T, 0, stream>>>(Wq, Wqb, (int)nW);
    k_cvt<<<(unsigned)((nW + T - 1) / T), T, 0, stream>>>(Wk, Wkb, (int)nW);
    k_cvt<<<(unsigned)((nW + T - 1) / T), T, 0, stream>>>(Wv, Wvb, (int)nW);
    k_cvt<<<(unsigned)((nW + T - 1) / T), T, 0, stream>>>(Wr, Wrb, (int)nW);
    k_cvt<<<(unsigned)((nW + T - 1) / T), T, 0, stream>>>(Wo, Wob, (int)nW);
    k_build_kv<<<(unsigned)(nKV / T), T, 0, stream>>>(mem, x, kvb);

    dim3 blk(256);
    // projections (all K = 1024)
    gemm_bf16nt<unsigned short><<<dim3(2048 / 128, 1024 / 64), blk, 0, stream>>>(xb,  Wqb, qwb, 2048, 1024, 1024);
    gemm_bf16nt<unsigned short><<<dim3(4096 / 128, 1024 / 64), blk, 0, stream>>>(kvb, Wkb, kwb, 4096, 1024, 1024);
    gemm_bf16nt<unsigned short><<<dim3(4096 / 128, 1024 / 64), blk, 0, stream>>>(kvb, Wvb, vwb, 4096, 1024, 1024);
    gemm_bf16nt<unsigned short><<<dim3(2048 / 128, 1024 / 64), blk, 0, stream>>>(rb,  Wrb, rwb, 2048, 1024, 1024);

    k_addbias<<<(unsigned)((nX + T - 1) / T), T, 0, stream>>>(qwb, ru, rv, qac, qbd, (int)nX);
    k_transpose_v<<<(unsigned)(nKV / T), T, 0, stream>>>(vwb, vtb);

    attn_relpos<<<dim3(B_ * H_ * (S_ / 128)), blk, 0, stream>>>(qac, qbd, kwb, rwb, vtb, msk, atb);

    // output projection -> fp32 result
    gemm_bf16nt<float><<<dim3(2048 / 128, 1024 / 64), blk, 0, stream>>>(atb, Wob, out, 2048, 1024, 1024);

    (void)in_sizes; (void)n_in; (void)out_size; (void)ws_size;
}
